// QuantumAttention_1365799600510
// MI455X (gfx1250) — matrix operand order repaired and checked by reading, not compiled, not run on hardware
//
#include <hip/hip_runtime.h>
#include <hip/hip_bf16.h>
#include <math.h>

typedef __attribute__((ext_vector_type(16))) _Float16 v16h;
typedef __attribute__((ext_vector_type(8)))  float    v8f;
typedef __attribute__((ext_vector_type(4)))  unsigned int v4u;
typedef __attribute__((ext_vector_type(8)))  int      v8i;
typedef __attribute__((ext_vector_type(4)))  int      v4i;

#define BQ 2
#define SEQ 2048
#define EMB 1024
#define NH 16
#define HD 64

// ---------------------------------------------------------------------------
// Tensor Data Mover: async 2D f16 tile load Global -> LDS.
// Builds a D# per CDNA5 ISA 8.3/8.4: group0 {count=1, lds_addr, global_addr(57b),
// type=2}; group1 {data_size=2B, tensor dims == tile dims (in-bounds tile),
// tile_dim0/1, dim0 stride}. Issue from one wave only; TDM ignores EXEC.
// Completion: s_wait_tensorcnt (tensor ops retire in order) + __syncthreads.
// ---------------------------------------------------------------------------
__device__ __forceinline__ void tdm_load_2d(const void* gptr, void* lptr,
                                            unsigned rows, unsigned row_elems,
                                            unsigned row_stride_elems) {
  unsigned long long ga = (unsigned long long)(uintptr_t)gptr;
  unsigned lds = (unsigned)(uintptr_t)lptr;   // generic LDS ptr: low 32 bits = LDS offset
  v4u g0;
  g0[0] = 1u;                                          // count=1, user mode
  g0[1] = lds;                                         // lds_addr
  g0[2] = (unsigned)(ga & 0xffffffffu);                // global_addr[31:0]
  g0[3] = (unsigned)((ga >> 32) & 0x1ffffffu) | (2u << 30);  // addr[56:32] | type=2
  v8i g1;
  g1[0] = (int)(1u << 16);                                         // data_size=1 (2 bytes)
  g1[1] = (int)((row_elems & 0xffffu) << 16);                      // tensor_dim0[15:0]
  g1[2] = (int)(((row_elems >> 16) & 0xffffu) | ((rows & 0xffffu) << 16)); // dim0 hi | dim1 lo
  g1[3] = (int)(((rows >> 16) & 0xffffu) | ((row_elems & 0xffffu) << 16)); // dim1 hi | tile_dim0
  g1[4] = (int)(rows & 0xffffu);                                   // tile_dim1 | tile_dim2=0
  g1[5] = (int)row_stride_elems;                                   // tensor_dim0_stride lo32
  g1[6] = 0;
  g1[7] = 0;
  v4i z = {0, 0, 0, 0};
#if defined(__clang_major__) && (__clang_major__ >= 23)
  v8i z8 = {0, 0, 0, 0, 0, 0, 0, 0};
  __builtin_amdgcn_tensor_load_to_lds(g0, g1, z, z, z8, 0);
#else
  __builtin_amdgcn_tensor_load_to_lds(g0, g1, z, z, 0);
#endif
}

// ---------------------------------------------------------------------------
// WMMA fragment loaders (wave32, f16 16x16x32); A and B both use k = 16*(lane/16) + s
// ---------------------------------------------------------------------------
__device__ __forceinline__ v16h load_a_f16(const _Float16* base, int pitch, int k0, int lane) {
  const int M  = lane & 15;
  const int kb = (lane >> 4) * 16;
  const _Float16* p = base + (size_t)M * pitch + k0 + kb;
  v16h r;
#pragma unroll
  for (int j = 0; j < 8; ++j) r[j] = p[j];
#pragma unroll
  for (int j = 0; j < 8; ++j) r[8 + j] = p[8 + j];
  return r;
}

__device__ __forceinline__ v16h load_b_t(const _Float16* bt, int pitch, int n0, int k0, int lane) {
  const _Float16* p = bt + (size_t)(n0 + (lane & 15)) * pitch + k0 + ((lane >> 4) * 16);
  v16h r;
#pragma unroll
  for (int j = 0; j < 16; ++j) r[j] = p[j];
  return r;
}

__device__ __forceinline__ v8f wmma_f16(v16h a, v16h b, v8f c) {
  return __builtin_amdgcn_wmma_f32_16x16x32_f16(false, a, false, b, (short)0, c, false, false);
}

// ---------------------------------------------------------------------------
__global__ void cvt_f32_f16(const float* __restrict__ src, _Float16* __restrict__ dst, int n) {
  int i = blockIdx.x * blockDim.x + threadIdx.x;
  if (i < n) dst[i] = (_Float16)src[i];
}

// ---------------------------------------------------------------------------
// f16 WMMA GEMM: C[M,N] = A[M,K]*B[K,N] + bias[N]; 64x64 tile / 256-thread WG.
// A tile staged by TDM; B tile transposed by threads. Per wave: 2 col tiles x
// 2 split accumulators -> 4 independent WMMAs per k-chunk (hides hazard NOPs).
// ---------------------------------------------------------------------------
template <bool F16OUT>
__global__ __launch_bounds__(256) void gemm_f16(const _Float16* __restrict__ A,
                                                const _Float16* __restrict__ B,
                                                const float* __restrict__ bias,
                                                void* __restrict__ C,
                                                int M, int N, int K) {
  __shared__ _Float16 As[64 * 64];
  __shared__ _Float16 Bt[64 * 64];  // Bt[n][k]

  const int tid  = threadIdx.x;
  const int lane = tid & 31;
  const int wave = tid >> 5;
  const int m0   = blockIdx.y * 64;
  const int n0b  = blockIdx.x * 64;

  const int mq  = (wave >> 1) * 16;
  const int nc0 = ((wave & 1) * 2) * 16;
  const int nc1 = nc0 + 16;

  v8f acc0a = {}, acc0b = {}, acc1a = {}, acc1b = {};

  const int ar = tid >> 2;
  const int ac = (tid & 3) * 16;

  for (int k0 = 0; k0 < K; k0 += 64) {
    __syncthreads();
    if (wave == 0)
      tdm_load_2d(A + (size_t)m0 * K + k0, As, 64, 64, (unsigned)K);
    {
      const _Float16* g = B + (size_t)(k0 + ar) * N + n0b + ac;
      if (k0 + 64 < K) __builtin_prefetch(g + (size_t)64 * N, 0, 0);
#pragma unroll
      for (int j = 0; j < 16; ++j) Bt[(ac + j) * 64 + ar] = g[j];
    }
    if (wave == 0) __builtin_amdgcn_s_wait_tensorcnt(0);
    __syncthreads();

    v16h a0  = load_a_f16(As + mq * 64, 64, 0, lane);
    v16h a1  = load_a_f16(As + mq * 64, 64, 32, lane);
    v16h b00 = load_b_t(Bt, 64, nc0, 0, lane);
    v16h b10 = load_b_t(Bt, 64, nc1, 0, lane);
    v16h b01 = load_b_t(Bt, 64, nc0, 32, lane);
    v16h b11 = load_b_t(Bt, 64, nc1, 32, lane);
    acc0a = wmma_f16(a0, b00, acc0a);   // 4 independent WMMAs
    acc1a = wmma_f16(a0, b10, acc1a);
    acc0b = wmma_f16(a1, b01, acc0b);
    acc1b = wmma_f16(a1, b11, acc1b);
  }

  const int col0 = n0b + nc0 + (lane & 15);
  const int col1 = n0b + nc1 + (lane & 15);
  const float bv0 = bias[col0];
  const float bv1 = bias[col1];
#pragma unroll
  for (int r = 0; r < 8; ++r) {
    const int row = m0 + mq + r + ((lane >> 4) * 8);
    const float v0 = acc0a[r] + acc0b[r] + bv0;
    const float v1 = acc1a[r] + acc1b[r] + bv1;
    if (F16OUT) {
      ((_Float16*)C)[(size_t)row * N + col0] = (_Float16)v0;
      ((_Float16*)C)[(size_t)row * N + col1] = (_Float16)v1;
    } else {
      ((float*)C)[(size_t)row * N + col0] = v0;
      ((float*)C)[(size_t)row * N + col1] = v1;
    }
  }
}

// ---------------------------------------------------------------------------
// FWHT-64 per wave32: lane holds x[d=lane] and x[d=lane+32]
// ---------------------------------------------------------------------------
__device__ __forceinline__ void fwht64(float& a, float& b, int lane) {
#pragma unroll
  for (int h = 1; h <= 16; h <<= 1) {
    float pa = __shfl_xor(a, h, 32);
    float pb = __shfl_xor(b, h, 32);
    a = (lane & h) ? (pa - a) : (a + pa);
    b = (lane & h) ? (pb - b) : (b + pb);
  }
  float na = a + b, nb = a - b;
  a = na * 0.125f;
  b = nb * 0.125f;
}

__global__ __launch_bounds__(256) void transform_kernel(const _Float16* __restrict__ qkv,
                                                        const float* __restrict__ diag,
                                                        const float* __restrict__ alphap,
                                                        const float* __restrict__ bias_h,
                                                        _Float16* __restrict__ qh,
                                                        _Float16* __restrict__ kh,
                                                        _Float16* __restrict__ vh) {
  const int lane = threadIdx.x & 31;
  const int wave = threadIdx.x >> 5;
  const int r = blockIdx.x * 8 + wave;
  const int h_ = r & (NH - 1);
  const int s_ = (r >> 4) & (SEQ - 1);
  const int b_ = r >> 15;

  const size_t base  = ((size_t)(b_ * SEQ + s_)) * (3 * EMB) + h_ * HD;
  const size_t hbase = ((size_t)((b_ * NH + h_) * SEQ + s_)) * HD;

  const float alpha = alphap[0];
  const float dv0 = diag[lane],   dv1 = diag[lane + 32];
  const float bh0 = bias_h[lane], bh1 = bias_h[lane + 32];

  vh[hbase + lane]      = qkv[base + 2 * EMB + lane];
  vh[hbase + lane + 32] = qkv[base + 2 * EMB + lane + 32];

#pragma unroll
  for (int t = 0; t < 2; ++t) {
    const size_t off = base + (size_t)t * EMB;
    const float x0 = (float)qkv[off + lane];
    const float x1 = (float)qkv[off + lane + 32];
    float a = x0, bb = x1;
    fwht64(a, bb, lane);
    a *= dv0; bb *= dv1;
    fwht64(a, bb, lane);
    const float y0 = x0 + alpha * a  + bh0;
    const float y1 = x1 + alpha * bb + bh1;
    _Float16* dst = (t == 0) ? qh : kh;
    dst[hbase + lane]      = (_Float16)y0;
    dst[hbase + lane + 32] = (_Float16)y1;
  }
}

// ---------------------------------------------------------------------------
// Attention: one WG = one (b,h) x 32-query block. 32x2048 f16 score block in
// LDS (legal on CDNA5: 320KB/WGP). Phase 1 double-buffers 128-key K tiles via
// TDM while waves run WMMAs. Per wave: 2 col tiles x 2 split accumulators.
// LDS halfs: Qs[2048] | K0[8192] | K1[8192] | S[65536] = 167,936 bytes.
// ---------------------------------------------------------------------------
__global__ __launch_bounds__(256) void attn_kernel(const _Float16* __restrict__ qh,
                                                   const _Float16* __restrict__ kh,
                                                   const _Float16* __restrict__ vh,
                                                   const int* __restrict__ mask,
                                                   _Float16* __restrict__ attn) {
  extern __shared__ _Float16 smem[];
  _Float16* Qs = smem;                       // 32*64
  _Float16* K0 = smem + 2048;                // 128*64
  _Float16* K1 = smem + 2048 + 8192;         // 128*64
  _Float16* S  = smem + 2048 + 2 * 8192;     // 32*2048

  const int tid  = threadIdx.x;
  const int lane = tid & 31;
  const int wave = tid >> 5;
  const int bh = blockIdx.y;
  const int b_ = bh >> 4;
  const int h_ = bh & 15;
  const int q0 = blockIdx.x * 32;
  const size_t headbase = (size_t)bh * SEQ * HD;

  const int mq  = (wave >> 2) * 16;   // 2 query sub-tiles
  const int ntA = (wave & 3) * 32;    // phase-1: 8 key sub-tiles, 2 per wave
  const int ntB = ntA + 16;
  const int nd  = (wave & 3) * 16;    // phase-3: 4 d sub-tiles

  const _Float16* kbase = kh + headbase;

  // ---- phase 1: S = Q K^T / 8, TDM double-buffered 128-key tiles ----
  if (wave == 0) {
    tdm_load_2d(qh + headbase + (size_t)q0 * HD, Qs, 32, 64, 64);
    tdm_load_2d(kbase, K0, 128, 64, 64);
  }
  for (int kt = 0; kt < SEQ / 128; ++kt) {
    if (wave == 0) {
      if (kt + 1 < SEQ / 128) {
        tdm_load_2d(kbase + (size_t)(kt + 1) * 128 * HD, ((kt + 1) & 1) ? K1 : K0,
                    128, 64, 64);
        __builtin_amdgcn_s_wait_tensorcnt(1);   // in-order: tile kt (and Q) done
      } else {
        __builtin_amdgcn_s_wait_tensorcnt(0);
      }
    }
    __syncthreads();
    const _Float16* Kt = (kt & 1) ? K1 : K0;

    v16h qf0 = load_a_f16(Qs + mq * 64, 64, 0, lane);
    v16h qf1 = load_a_f16(Qs + mq * 64, 64, 32, lane);
    v16h bA0 = load_b_t(Kt, 64, ntA, 0, lane);
    v16h bB0 = load_b_t(Kt, 64, ntB, 0, lane);
    v16h bA1 = load_b_t(Kt, 64, ntA, 32, lane);
    v16h bB1 = load_b_t(Kt, 64, ntB, 32, lane);
    v8f aA0 = {}, aA1 = {}, aB0 = {}, aB1 = {};
    aA0 = wmma_f16(qf0, bA0, aA0);
    aB0 = wmma_f16(qf0, bB0, aB0);
    aA1 = wmma_f16(qf1, bA1, aA1);
    aB1 = wmma_f16(qf1, bB1, aB1);

    const int colA = kt * 128 + ntA + (lane & 15);
    const int colB = kt * 128 + ntB + (lane & 15);
#pragma unroll
    for (int rr = 0; rr < 8; ++rr) {
      const int row = mq + rr + ((lane >> 4) * 8);
      S[row * SEQ + colA] = (_Float16)((aA0[rr] + aA1[rr]) * 0.125f);
      S[row * SEQ + colB] = (_Float16)((aB0[rr] + aB1[rr]) * 0.125f);
    }
    __syncthreads();  // tile fully consumed before TDM re-targets this buffer
  }

  // ---- phase 2: softmax over 2048 keys, 4 rows per wave ----
  const int* mrow = mask + b_ * SEQ;
  for (int rw = 0; rw < 4; ++rw) {
    _Float16* srow = S + (wave * 4 + rw) * SEQ;
    float mx = -3.0e38f;
    for (int c = lane; c < SEQ; c += 32) {
      const float v = (float)srow[c];
      if (mrow[c] != 0) mx = fmaxf(mx, v);
    }
#pragma unroll
    for (int h = 1; h < 32; h <<= 1) mx = fmaxf(mx, __shfl_xor(mx, h, 32));
    float sum = 0.0f;
    for (int c = lane; c < SEQ; c += 32) {
      const float v = (float)srow[c];
      const float p = (mrow[c] != 0) ? __expf(v - mx) : 0.0f;
      srow[c] = (_Float16)p;
      sum += p;
    }
#pragma unroll
    for (int h = 1; h < 32; h <<= 1) sum += __shfl_xor(sum, h, 32);
    const float inv = 1.0f / sum;
    for (int c = lane; c < SEQ; c += 32)
      srow[c] = (_Float16)((float)srow[c] * inv);
  }

  // ---- phase 3: O = P V (split accumulators; V^T staged by threads) ----
  v8f o0 = {}, o1 = {};
  for (int kt = 0; kt < SEQ / 64; ++kt) {
    __syncthreads();
    {
      const int idx = tid * 16;
      const int key = idx >> 6;
      const int dc  = idx & 63;
      const _Float16* g = vh + headbase + (size_t)(kt * 64 + key) * HD + dc;
#pragma unroll
      for (int j = 0; j < 16; ++j) K0[(dc + j) * 64 + key] = g[j];  // V^T[d][key]
    }
    __syncthreads();
    v16h a0 = load_a_f16(S + mq * SEQ, SEQ, kt * 64, lane);
    v16h a1 = load_a_f16(S + mq * SEQ, SEQ, kt * 64 + 32, lane);
    v16h b0 = load_b_t(K0, 64, nd, 0, lane);
    v16h b1 = load_b_t(K0, 64, nd, 32, lane);
    o0 = wmma_f16(a0, b0, o0);   // independent chains
    o1 = wmma_f16(a1, b1, o1);
  }

  const int colg = h_ * HD + nd + (lane & 15);
#pragma unroll
  for (int rr = 0; rr < 8; ++rr) {
    const int row = q0 + mq + rr + ((lane >> 4) * 8);
    attn[((size_t)(b_ * SEQ + row)) * EMB + colg] = (_Float16)(o0[rr] + o1[rr]);
  }
}

// ---------------------------------------------------------------------------
extern "C" void kernel_launch(void* const* d_in, const int* in_sizes, int n_in,
                              void* d_out, int out_size, void* d_ws, size_t ws_size,
                              hipStream_t stream) {
  const float* x     = (const float*)d_in[0];
  const int*   mask  = (const int*)d_in[1];
  const float* Wqkv  = (const float*)d_in[2];
  const float* bqkv  = (const float*)d_in[3];
  const float* Wout  = (const float*)d_in[4];
  const float* bout  = (const float*)d_in[5];
  const float* diag  = (const float*)d_in[6];
  const float* alpha = (const float*)d_in[7];
  const float* biash = (const float*)d_in[8];
  float* out = (float*)d_out;

  char* ws = (char*)d_ws;
  const size_t MB = 1024 * 1024;
  _Float16* xh    = (_Float16*)(ws);            //  8 MB  [4096][1024]
  _Float16* wqkvh = (_Float16*)(ws + 8 * MB);   //  6 MB  [1024][3072]
  _Float16* wouth = (_Float16*)(ws + 14 * MB);  //  2 MB  [1024][1024]
  _Float16* qkvh  = (_Float16*)(ws + 16 * MB);  // 24 MB  [4096][3072]
  _Float16* qhp   = (_Float16*)(ws + 40 * MB);  //  8 MB  [b,H,s,D]
  _Float16* khp   = (_Float16*)(ws + 48 * MB);  //  8 MB
  _Float16* vhp   = (_Float16*)(ws + 56 * MB);  //  8 MB
  _Float16* attnh = (_Float16*)(ws + 64 * MB);  //  8 MB  [4096][1024]

  const int nX  = BQ * SEQ * EMB;
  const int nWq = EMB * 3 * EMB;
  const int nWo = EMB * EMB;
  cvt_f32_f16<<<(nX  + 255) / 256, 256, 0, stream>>>(x,    xh,    nX);
  cvt_f32_f16<<<(nWq + 255) / 256, 256, 0, stream>>>(Wqkv, wqkvh, nWq);
  cvt_f32_f16<<<(nWo + 255) / 256, 256, 0, stream>>>(Wout, wouth, nWo);

  gemm_f16<true><<<dim3(3 * EMB / 64, BQ * SEQ / 64), 256, 0, stream>>>(
      xh, wqkvh, bqkv, qkvh, BQ * SEQ, 3 * EMB, EMB);

  transform_kernel<<<(BQ * SEQ * NH) / 8, 256, 0, stream>>>(
      qkvh, diag, alpha, biash, qhp, khp, vhp);

  const size_t smem = (size_t)(2048 + 2 * 8192 + 32 * SEQ) * sizeof(_Float16);  // 167,936 B
  attn_kernel<<<dim3(SEQ / 32, BQ * NH), 256, smem, stream>>>(qhp, khp, vhp, mask, attnh);

  gemm_f16<false><<<dim3(EMB / 64, BQ * SEQ / 64), 256, 0, stream>>>(
      attnh, wouth, bout, out, BQ * SEQ, EMB, EMB);
}
